// Interaction_88880053223889
// MI455X (gfx1250) — matrix operand order repaired and checked by reading, not compiled, not run on hardware
//
#include <hip/hip_runtime.h>

// SchNet cfconv interaction, fused, WMMA f16 (f32 accumulate) + TDM async x-staging.
// MI455X / gfx1250, wave32.
#define Bsz 4
#define Nn 512
#define Ff 128
#define Kk 32
#define MSPLIT 4
#define MCHUNK (Nn / MSPLIT) /* 128 m-values per workgroup */
#define XSTRIDE 132          /* 128 dwords + 4 dwords TDM pad per row */
#define GAMMA_ 10.0f

typedef __attribute__((ext_vector_type(16))) _Float16 v16h;
typedef __attribute__((ext_vector_type(8)))  _Float16 v8h;
typedef __attribute__((ext_vector_type(8)))  float    v8f;
typedef __attribute__((ext_vector_type(4)))  unsigned int u32x4;
typedef __attribute__((ext_vector_type(8)))  int      i32x8;
typedef __attribute__((ext_vector_type(4)))  int      i32x4;

// shifted softplus: ln(0.5 + 0.5*e^v); inputs here are O(1), no overflow guard needed
__device__ __forceinline__ float sspf(float v) {
    return __logf(fmaf(0.5f, __expf(v), 0.5f));
}

// 16-bit A-matrix element->K mapping (ISA 7.12.2): e<8 -> K=e+8*hi ; e>=8 -> K=16+(e-8)+8*hi
__device__ __forceinline__ int kA(int e, int hi) { return e + (e & 8) + 8 * hi; }

__device__ __forceinline__ v8f wmma16(v16h a, v16h b, v8f c) {
    return __builtin_amdgcn_wmma_f32_16x16x32_f16(false, a, false, b, (short)0, c,
                                                  false, false);
}
// same-A follow-up WMMA: set reuse-A operand hint (ISA OPSEL RA)
__device__ __forceinline__ v8f wmma16_ra(v16h a, v16h b, v8f c) {
    return __builtin_amdgcn_wmma_f32_16x16x32_f16(false, a, false, b, (short)0, c,
                                                  true, false);
}

__device__ __forceinline__ v16h cat8(v8h lo, v8h hi) {
    return __builtin_shufflevector(lo, hi, 0, 1, 2, 3, 4, 5, 6, 7,
                                   8, 9, 10, 11, 12, 13, 14, 15);
}

// ---------------- kernel 1: out = atom_emb ----------------
__global__ __launch_bounds__(256) void init_out_kernel(const float* __restrict__ emb,
                                                       float* __restrict__ out) {
    int i = blockIdx.x * blockDim.x + threadIdx.x; // one float4 each
    const float4* s = (const float4*)emb;
    float4* d = (float4*)out;
    d[i] = s[i];
}

// ---------------- kernel 2: x = emb @ Ww^T + Wb (WMMA) ----------------
__global__ __launch_bounds__(256) void x_gemm_kernel(const float* __restrict__ emb,
                                                     const float* __restrict__ Ww,
                                                     const float* __restrict__ Wb,
                                                     float* __restrict__ x) {
    const int lane = threadIdx.x & 31, wave = threadIdx.x >> 5;
    const int hi = lane >> 4, l16 = lane & 15;
    const int tile = blockIdx.x * 8 + wave; // 0..127 row tiles of 16 (flat b*N+m)
    const int row0 = tile * 16;

    v8f acc[8];
#pragma unroll
    for (int t = 0; t < 8; ++t) {
        float bias = Wb[16 * t + l16];
#pragma unroll
        for (int v = 0; v < 8; ++v) acc[t][v] = bias;
    }
#pragma unroll
    for (int kk = 0; kk < 4; ++kk) {
        v16h a;
        const float* ar = emb + (size_t)(row0 + l16) * Ff + 32 * kk;
#pragma unroll
        for (int e = 0; e < 16; ++e) a[e] = (_Float16)ar[16 * hi + e];
#pragma unroll
        for (int t = 0; t < 8; ++t) {
            // B[k=g][col=f] = Ww[f][g]; 16 consecutive g for fixed f
            const float* br = Ww + (size_t)(16 * t + l16) * Ff + 32 * kk + 16 * hi;
            v16h bf;
#pragma unroll
            for (int e = 0; e < 16; ++e) bf[e] = (_Float16)br[e];
            acc[t] = (t == 0) ? wmma16(a, bf, acc[t]) : wmma16_ra(a, bf, acc[t]);
        }
    }
#pragma unroll
    for (int t = 0; t < 8; ++t)
#pragma unroll
        for (int v = 0; v < 8; ++v)
            x[(size_t)(row0 + v + 8 * hi) * Ff + 16 * t + l16] = acc[t][v];
}

// ---------------- kernel 3: fused RBF -> GEMM1 -> ssp -> GEMM2 -> ssp -> mask
//                            -> per-channel contraction; x staged by TDM ----------------
__global__ __launch_bounds__(256, 1) void cfconv_kernel(
    const float* __restrict__ coords, const float* __restrict__ centers,
    const float* __restrict__ w1, const float* __restrict__ b1g,
    const float* __restrict__ w2, const float* __restrict__ b2g,
    const float* __restrict__ x, float* __restrict__ out) {
    __shared__ _Float16 sW2t[Ff * 136];        // [g_out][f_in], 16B-aligned rows
    __shared__ float    sX[2][16 * XSTRIDE];   // TDM double buffer: [m][128 f + 4 pad]
    __shared__ _Float16 sH[8][16 * 40];        // per-wave C->A staging

    const int tid = threadIdx.x;
    const int lane = tid & 31, wave = tid >> 5;
    const int hi = lane >> 4, l16 = lane & 15;
    const int b = blockIdx.y;
    const int nb = blockIdx.x * 16;
    const int mstart = blockIdx.z * MCHUNK;

    // stage w2^T into LDS once
#pragma unroll 4
    for (int r = 0; r < 64; ++r) {
        int idx = tid + 256 * r; // coalesced read of w2[f_in][g]
        int fin = idx >> 7, g = idx & 127;
        sW2t[g * 136 + fin] = (_Float16)w2[idx];
    }

    // ---- TDM descriptor (2D tile, f32, row pad 128->132 dwords) ----
    // group1: data_size=4B(2), pad_enable, pad_interval=128dw(6), pad_amount=4dw(3),
    //         tensor_dim0=128, tensor_dim1=16, tile_dim0=128, tile_dim1=16, stride0=128
    const i32x8 tg1 = {0x07920000, (int)(128u << 16), (int)(16u << 16),
                       (int)(128u << 16), 16, 128, 0, 0};
    const i32x4 tz4 = {0, 0, 0, 0};
    const i32x8 tz8 = {0, 0, 0, 0, 0, 0, 0, 0};
    const unsigned int ldsX0 = (unsigned int)(size_t)(&sX[0][0]);
    const unsigned int ldsX1 = (unsigned int)(size_t)(&sX[1][0]);

    // per-lane constants
    float cen[16];
#pragma unroll
    for (int e = 0; e < 16; ++e) cen[e] = centers[kA(e, hi)];
    float bb1[8], bb2[8];
#pragma unroll
    for (int t = 0; t < 8; ++t) {
        bb1[t] = b1g[16 * t + l16];
        bb2[t] = b2g[16 * t + l16];
    }
    // w1 B-fragments (K=32 = one WMMA k-step), kept in registers
    v16h w1f[8];
#pragma unroll
    for (int t = 0; t < 8; ++t)
#pragma unroll
        for (int e = 0; e < 16; ++e)
            w1f[t][e] = (_Float16)w1[kA(e, hi) * Ff + 16 * t + l16];

    float oacc[2][8];
#pragma unroll
    for (int j = 0; j < 2; ++j)
#pragma unroll
        for (int t = 0; t < 8; ++t) oacc[j][t] = 0.0f;

    // prologue: DMA first x chunk into buffer 0 (wave 0 issues, TENSORcnt tracks)
    if (wave == 0) {
        unsigned long long ga =
            (unsigned long long)(size_t)(x + ((size_t)b * Nn + mstart) * Ff);
        u32x4 g0;
        g0[0] = 1u; // count=1 valid descriptor
        g0[1] = ldsX0;
        g0[2] = (unsigned int)ga;
        g0[3] = (unsigned int)((ga >> 32) & 0x01FFFFFFu) | 0x80000000u; // type=2
        __builtin_amdgcn_tensor_load_to_lds(g0, tg1, tz4, tz4, tz8, 0);
    }

    const float* cb = coords + (size_t)b * Nn * 3;
    _Float16* sHw = &sH[wave][0];

    for (int mc = 0; mc < MCHUNK; mc += 16) {
        const int mbase = mstart + mc;
        const int cur = (mc >> 4) & 1;
        if (wave == 0) {
            __builtin_amdgcn_s_wait_tensorcnt(0); // current buffer DMA complete
            if (mc + 16 < MCHUNK) {               // prefetch next chunk (overlaps compute)
                unsigned long long ga = (unsigned long long)(size_t)(
                    x + ((size_t)b * Nn + mbase + 16) * Ff);
                u32x4 g0;
                g0[0] = 1u;
                g0[1] = cur ? ldsX0 : ldsX1;
                g0[2] = (unsigned int)ga;
                g0[3] = (unsigned int)((ga >> 32) & 0x01FFFFFFu) | 0x80000000u;
                __builtin_amdgcn_tensor_load_to_lds(g0, tg1, tz4, tz4, tz8, 0);
            }
        }
        __syncthreads(); // current buffer visible to all waves
        const float* xbuf = &sX[cur][0];

        const int m = mbase + l16; // this lane's matrix row = neighbor index
        const float mx = cb[m * 3 + 0], my = cb[m * 3 + 1], mz = cb[m * 3 + 2];

#pragma unroll
        for (int j = 0; j < 2; ++j) {
            const int n = nb + wave + 8 * j;
            const float dx = cb[n * 3 + 0] - mx;
            const float dy = cb[n * 3 + 1] - my;
            const float dz = cb[n * 3 + 2] - mz;
            const float d = sqrtf(dx * dx + dy * dy + dz * dz + 1e-12f);

            // RBF A-fragment: rows = m (16), K = 32 gaussians
            v16h a1;
#pragma unroll
            for (int e = 0; e < 16; ++e) {
                float u = d - cen[e];
                a1[e] = (_Float16)__expf(-GAMMA_ * u * u);
            }

            // GEMM1 (16x128 <- 16x32 * 32x128), bias in C operand, then ssp
            v8f h[8];
#pragma unroll
            for (int t = 0; t < 8; ++t) {
                v8f c;
#pragma unroll
                for (int v = 0; v < 8; ++v) c[v] = bb1[t];
                h[t] = (t == 0) ? wmma16(a1, w1f[t], c) : wmma16_ra(a1, w1f[t], c);
            }
#pragma unroll
            for (int t = 0; t < 8; ++t)
#pragma unroll
                for (int v = 0; v < 8; ++v) h[t][v] = sspf(h[t][v]);

            // GEMM2 (16x128 <- 16x128 * 128x128), C->A relayout via per-wave LDS
            v8f acc[8];
#pragma unroll
            for (int t = 0; t < 8; ++t) {
                v8f c;
#pragma unroll
                for (int v = 0; v < 8; ++v) c[v] = bb2[t];
                acc[t] = c;
            }
#pragma unroll
            for (int kk = 0; kk < 4; ++kk) {
#pragma unroll
                for (int th = 0; th < 2; ++th) {
                    int t2 = 2 * kk + th;
#pragma unroll
                    for (int v = 0; v < 8; ++v)
                        sHw[(v + 8 * hi) * 40 + 16 * th + l16] = (_Float16)h[t2][v];
                }
                const v8h* plo = (const v8h*)&sHw[l16 * 40 + 16 * hi];
                const v8h* phi = (const v8h*)&sHw[l16 * 40 + 16 * hi + 8];
                v16h a2 = cat8(*plo, *phi);
#pragma unroll
                for (int t = 0; t < 8; ++t) {
                    const v8h* q =
                        (const v8h*)&sW2t[(16 * t + l16) * 136 + 32 * kk + 16 * hi];
                    v16h bf = cat8(q[0], q[1]);
                    acc[t] = (t == 0) ? wmma16(a2, bf, acc[t])
                                      : wmma16_ra(a2, bf, acc[t]);
                }
            }

            // ssp + zero self-interaction row (m == n)
            const int dr = n - mbase; // valid row in [0,16)
#pragma unroll
            for (int t = 0; t < 8; ++t)
#pragma unroll
                for (int v = 0; v < 8; ++v) {
                    float w = sspf(acc[t][v]);
                    if (v + 8 * hi == dr) w = 0.0f;
                    acc[t][v] = w;
                }

            // per-channel contraction: out[n,f] += sum_row Wf[row,f]*x[mbase+row,f]
            const float* xr = xbuf + 8 * hi * XSTRIDE + l16;
#pragma unroll
            for (int t = 0; t < 8; ++t) {
                float p = 0.0f;
#pragma unroll
                for (int v = 0; v < 8; ++v) p += acc[t][v] * xr[v * XSTRIDE + 16 * t];
                p += __shfl_xor(p, 16, 32); // combine row halves (lanes L <-> L+16)
                oacc[j][t] += p;
            }
        }
        __syncthreads(); // all reads of this buffer done before TDM reuses it
    }

    // accumulate partial sums (MSPLIT workgroups per n) into out
    if (lane < 16) {
#pragma unroll
        for (int j = 0; j < 2; ++j) {
            const int n = nb + wave + 8 * j;
            float* op = out + ((size_t)b * Nn + n) * Ff;
#pragma unroll
            for (int t = 0; t < 8; ++t) atomicAdd(&op[16 * t + l16], oacc[j][t]);
        }
    }
}

extern "C" void kernel_launch(void* const* d_in, const int* in_sizes, int n_in,
                              void* d_out, int out_size, void* d_ws, size_t ws_size,
                              hipStream_t stream) {
    const float* coords = (const float*)d_in[0];
    const float* emb    = (const float*)d_in[1];
    const float* cent   = (const float*)d_in[2];
    const float* w1     = (const float*)d_in[3];
    const float* b1     = (const float*)d_in[4];
    const float* w2     = (const float*)d_in[5];
    const float* b2     = (const float*)d_in[6];
    const float* Ww     = (const float*)d_in[7];
    const float* Wb     = (const float*)d_in[8];
    float* out = (float*)d_out;
    float* x   = (float*)d_ws; // B*N*F fp32 = 1 MB scratch

    init_out_kernel<<<(Bsz * Nn * Ff / 4) / 256, 256, 0, stream>>>(emb, out);
    x_gemm_kernel<<<(Bsz * Nn / 16) / 8, 256, 0, stream>>>(emb, Ww, Wb, x);
    dim3 grid(Nn / 16, Bsz, MSPLIT);
    cfconv_kernel<<<grid, 256, 0, stream>>>(coords, cent, w1, b1, w2, b2, x, out);
}
